// GQAAttention_79783312491025
// MI455X (gfx1250) — matrix operand order repaired and checked by reading, not compiled, not run on hardware
//
#include <hip/hip_runtime.h>
#include <hip/hip_bf16.h>

// MI455X / gfx1250 GQA prefill. f16 operands (converted once), f32 WMMA accum.
// ws layout (bytes):
//   qkv_f32 : 0         .. 64MB   (8192 x 2048 f32)
//   qkv_h   : 64MB      .. +32MB  (8192 x 2048 f16)
//   x_h     : 96MB      .. +24MB  (8192 x 1536 f16)
//   attn_h  : 120MB     .. +24MB  (8192 x 1536 f16)
//   wqkv_h  : 144MB     .. +6MB   (2048 x 1536 f16)
//   wo_h    : 150MB     .. +4.5MB (1536 x 1536 f16)

typedef __attribute__((ext_vector_type(16))) _Float16 v16h;
typedef __attribute__((ext_vector_type(8)))  _Float16 v8h;
typedef __attribute__((ext_vector_type(4)))  _Float16 v4h;
typedef __attribute__((ext_vector_type(8)))  float    v8f;
typedef __attribute__((ext_vector_type(4)))  float    v4f;

#define HH    12
#define HKV_  2
#define DD    128
#define HID_  1536
#define NG    6
#define SS    2048
#define BB    4
#define QKV_N 2048
#define TOK   (BB*SS)
#define NSLOTS 16384

#define CAT16(lo, hi) __builtin_shufflevector(lo, hi, 0,1,2,3,4,5,6,7,8,9,10,11,12,13,14,15)

// A-fragment (16x32 f16) from row-major f16 matrix: lane l -> row m=l&15,
// elements 0..15 -> K=e+16*half (contiguous).
__device__ __forceinline__ v16h load_a_h(const _Float16* base, int ld,
                                         int row0, int k0, int lane) {
  int m = lane & 15, half = lane >> 4;
  const _Float16* p = base + (long long)(row0 + m) * ld + k0;
  v8h lo = *(const v8h*)(p + 16 * half);
  v8h hi = *(const v8h*)(p + 16 * half + 8);
  return CAT16(lo, hi);
}

// ---------------------------------------------------------------------------
// f32 -> f16 bulk conversion (vectorized: b128 load, 2x cvt_pk, b64 store)
// ---------------------------------------------------------------------------
__global__ __launch_bounds__(256)
void cvt_f16_kernel(const float* __restrict__ src, _Float16* __restrict__ dst,
                    long long n4) {
  long long i = (long long)blockIdx.x * blockDim.x + threadIdx.x;
  if (i >= n4) return;
  v4f s = ((const v4f*)src)[i];
  v4h d;
#pragma unroll
  for (int j = 0; j < 4; ++j) d[j] = (_Float16)s[j];
  ((v4h*)dst)[i] = d;
}

// ---------------------------------------------------------------------------
// GEMM: Y[M,N](f32) = Xh[M,K](f16) * Wh[N,K]^T(f16) (+bias).
// 32x64 macro-tile per wave: 2 A-frags x 4 B-frags -> 8 WMMAs per K-step
// (1.5 b128 loads per WMMA instead of 4).
// ---------------------------------------------------------------------------
__global__ __launch_bounds__(256)
void gemm_h_kernel(const _Float16* __restrict__ Xh, const _Float16* __restrict__ Wh,
                   const float* __restrict__ bias, float* __restrict__ Y,
                   int M, int N, int K) {
  int wave = threadIdx.x >> 5;
  int lane = threadIdx.x & 31;
  int row0 = blockIdx.x * 32;
  int n0   = (blockIdx.y * 8 + wave) * 64;
  if (row0 >= M || n0 >= N) return;
  int n = lane & 15, half = lane >> 4;

  const _Float16* ap0 = Xh + (long long)(row0 + (lane & 15)) * K;
  const _Float16* wp[4];
#pragma unroll
  for (int j = 0; j < 4; ++j)
    wp[j] = Wh + (long long)(n0 + j * 16 + n) * K + 16 * half;

  v8f acc[2][4];
#pragma unroll
  for (int mt = 0; mt < 2; ++mt)
#pragma unroll
    for (int j = 0; j < 4; ++j)
#pragma unroll
      for (int i = 0; i < 8; ++i) acc[mt][j][i] = 0.f;

  for (int k0 = 0; k0 < K; k0 += 32) {
    v16h a[2];
#pragma unroll
    for (int mt = 0; mt < 2; ++mt)
      a[mt] = load_a_h(Xh, K, row0 + mt * 16, k0, lane);
    __builtin_prefetch(ap0 + k0 + 32, 0, 0);
#pragma unroll
    for (int j = 0; j < 4; ++j) {
      v8h blo = *(const v8h*)(wp[j] + k0);
      v8h bhi = *(const v8h*)(wp[j] + k0 + 8);
      v16h b = CAT16(blo, bhi);
      __builtin_prefetch(wp[j] + k0 + 32, 0, 0);
#pragma unroll
      for (int mt = 0; mt < 2; ++mt)
        acc[mt][j] = __builtin_amdgcn_wmma_f32_16x16x32_f16(
            false, a[mt], false, b, (short)0, acc[mt][j], false, false);
    }
  }

#pragma unroll
  for (int j = 0; j < 4; ++j) {
    float bv = bias ? bias[n0 + j * 16 + n] : 0.f;
#pragma unroll
    for (int mt = 0; mt < 2; ++mt)
#pragma unroll
      for (int r = 0; r < 8; ++r) {
        int row = row0 + mt * 16 + r + 8 * half;
        Y[(long long)row * N + n0 + j * 16 + n] = acc[mt][j][r] + bv;
      }
  }
}

// ---------------------------------------------------------------------------
// RoPE in place on q,k (f32 qkv), scatter k/v to caches, emit f16 qkv copy.
// ---------------------------------------------------------------------------
__global__ __launch_bounds__(128)
void rope_scatter_kernel(float* __restrict__ qkv, _Float16* __restrict__ qkv_h,
                         const float* __restrict__ cosb,
                         const float* __restrict__ sinb,
                         const int* __restrict__ slot_map,
                         float* __restrict__ kc, float* __restrict__ vc) {
  int t   = blockIdx.x;
  int tid = threadIdx.x;
  float* row = qkv + (long long)t * QKV_N;
  const float* cp = cosb + (long long)t * DD;
  const float* sp = sinb + (long long)t * DD;

  int d = tid & 63;
  int group = tid >> 6;
  float c0 = cp[d], s0 = sp[d], c1 = cp[d + 64], s1 = sp[d + 64];
  for (int h = group; h < HH + HKV_; h += 2) {
    float* hp = (h < HH) ? (row + h * DD) : (row + HID_ + (h - HH) * DD);
    float x1 = hp[d], x2 = hp[d + 64];
    hp[d]      = x1 * c0 - x2 * s0;
    hp[d + 64] = x2 * c1 + x1 * s1;
  }
  __syncthreads();

  int slot = slot_map[t];
  for (int i = tid; i < HKV_ * DD; i += 128) {
    kc[(long long)slot * (HKV_ * DD) + i] = row[HID_ + i];
    vc[(long long)slot * (HKV_ * DD) + i] = row[HID_ + HKV_ * DD + i];
  }
  // f16 copy of the (roped) row for the attention kernel
  _Float16* hrow = qkv_h + (long long)t * QKV_N;
#pragma unroll
  for (int j = 0; j < 4; ++j) {
    int i = (tid * 4 + j * 512);
    v4f s = *(const v4f*)(row + i);
    v4h dh;
#pragma unroll
    for (int e = 0; e < 4; ++e) dh[e] = (_Float16)s[e];
    *(v4h*)(hrow + i) = dh;
  }
}

// ---------------------------------------------------------------------------
// Flash attention: one (b,h,16-row q-tile) per wave, 32-key tiles, f16 qkv.
// V tiles staged global->LDS with gfx1250 async loads (ASYNCcnt).
// ---------------------------------------------------------------------------
__global__ __launch_bounds__(128)
void flash_attn_kernel(const _Float16* __restrict__ qkv_h,
                       _Float16* __restrict__ O) {
  __shared__ _Float16 plds[4][16 * 32];      // P transpose staging
  __shared__ _Float16 vstage[4][32 * 128];   // V tile [key][d]
  int wave = threadIdx.x >> 5, lane = threadIdx.x & 31;
  int gw = blockIdx.x * 4 + wave;            // 0..6143
  int qt = gw & 127;
  int h  = (gw >> 7) % HH;
  int b  = gw / (128 * HH);
  int hk = h / NG;
  long long tb = (long long)b * SS;
  int n = lane & 15, half = lane >> 4;

  v16h qa[4];
#pragma unroll
  for (int c = 0; c < 4; ++c)
    qa[c] = load_a_h(qkv_h + tb * QKV_N + h * DD, QKV_N, qt * 16, c * 32, lane);

  float mrow[8], lrow[8];
  v8f oacc[8];
#pragma unroll
  for (int r = 0; r < 8; ++r) { mrow[r] = -1e30f; lrow[r] = 0.f; }
#pragma unroll
  for (int ch = 0; ch < 8; ++ch)
#pragma unroll
    for (int r = 0; r < 8; ++r) oacc[ch][r] = 0.f;

  const float scale = 0.08838834764831845f;  // 1/sqrt(128)
  int qend = qt * 16 + 15;
  _Float16* pl = plds[wave];
  _Float16* vt = vstage[wave];
  unsigned vt_lds = (unsigned)(uintptr_t)vt;                 // LDS byte offset
  unsigned lbase = vt_lds + (unsigned)(half * 256 + n * 16); // [key][d] rows
  // global base for this lane's V chunks: key = kt0 + half + 2i, 16B per chunk
  unsigned long long gvbase0 =
      (unsigned long long)(uintptr_t)qkv_h +
      (unsigned long long)(HID_ + HKV_ * DD + hk * DD) * 2 + (unsigned)(n * 16);

  for (int kt0 = 0; kt0 <= qend; kt0 += 32) {
    // ---- stage V tile (32 keys x 128 f16 = 8KB) via async loads ----
    asm volatile("s_wait_dscnt 0x0" ::: "memory");
    unsigned long long gvb = gvbase0 + (unsigned long long)(tb + kt0 + half) * (QKV_N * 2);
#pragma unroll
    for (int i = 0; i < 16; ++i) {
      unsigned long long ga = gvb + (unsigned long long)i * 8192u;
      unsigned la = lbase + (unsigned)i * 512u;
      asm volatile("global_load_async_to_lds_b128 %0, %1, off"
                   :: "v"(la), "v"(ga) : "memory");
    }

    // ---- scores S = Q K^T (overlaps with async V staging) ----
    v8f sa[2];
#pragma unroll
    for (int nt = 0; nt < 2; ++nt)
#pragma unroll
      for (int r = 0; r < 8; ++r) sa[nt][r] = 0.f;

#pragma unroll
    for (int nt = 0; nt < 2; ++nt) {
      int key = kt0 + nt * 16 + n;
      const _Float16* kp = qkv_h + (tb + key) * QKV_N + HID_ + hk * DD + 16 * half;
#pragma unroll
      for (int c = 0; c < 4; ++c) {
        v8h klo = *(const v8h*)(kp + c * 32);
        v8h khi = *(const v8h*)(kp + c * 32 + 8);
        v16h kb = CAT16(klo, khi);
        sa[nt] = __builtin_amdgcn_wmma_f32_16x16x32_f16(false, qa[c], false, kb,
                                                        (short)0, sa[nt], false, false);
      }
    }

    // ---- online softmax update ----
#pragma unroll
    for (int r = 0; r < 8; ++r) {
      int qrow = qt * 16 + r + 8 * half;
      float rmax = -1e30f;
#pragma unroll
      for (int nt = 0; nt < 2; ++nt) {
        int kk = kt0 + nt * 16 + n;
        float sv = sa[nt][r] * scale;
        if (kk > qrow) sv = -1e30f;
        sa[nt][r] = sv;
        rmax = fmaxf(rmax, sv);
      }
#pragma unroll
      for (int off = 1; off < 16; off <<= 1)
        rmax = fmaxf(rmax, __shfl_xor(rmax, off, 32));
      float mnew = fmaxf(mrow[r], rmax);
      float corr = __expf(mrow[r] - mnew);
      float rsum = 0.f;
#pragma unroll
      for (int nt = 0; nt < 2; ++nt) {
        float p = __expf(sa[nt][r] - mnew);
        sa[nt][r] = p;
        rsum += p;
      }
#pragma unroll
      for (int off = 1; off < 16; off <<= 1)
        rsum += __shfl_xor(rsum, off, 32);
      lrow[r] = lrow[r] * corr + rsum;
      mrow[r] = mnew;
#pragma unroll
      for (int ch = 0; ch < 8; ++ch) oacc[ch][r] *= corr;
    }

    // ---- P: C/D layout -> LDS -> A-fragment layout ----
#pragma unroll
    for (int nt = 0; nt < 2; ++nt)
#pragma unroll
      for (int r = 0; r < 8; ++r)
        pl[(r + 8 * half) * 32 + nt * 16 + n] = (_Float16)sa[nt][r];

    v16h pa;
#pragma unroll
    for (int e = 0; e < 8; ++e) pa[e]     = pl[n * 32 + e + 16 * half];
#pragma unroll
    for (int e = 0; e < 8; ++e) pa[8 + e] = pl[n * 32 + 8 + e + 16 * half];

    // ---- O += P V  (V B-fragments from the staged LDS tile) ----
    asm volatile("s_wait_asynccnt 0x0" ::: "memory");
#pragma unroll
    for (int ch = 0; ch < 8; ++ch) {
      v16h vb;
#pragma unroll
      for (int e = 0; e < 16; ++e)
        vb[e] = vt[(e + 16 * half) * 128 + ch * 16 + n];
      oacc[ch] = __builtin_amdgcn_wmma_f32_16x16x32_f16(false, pa, false, vb,
                                                        (short)0, oacc[ch], false, false);
    }
  }

  // ---- epilogue: normalize (rcp + mul, not div), store f16 for Wo GEMM ----
  float rinv[8];
#pragma unroll
  for (int r = 0; r < 8; ++r) rinv[r] = __builtin_amdgcn_rcpf(lrow[r]);
#pragma unroll
  for (int ch = 0; ch < 8; ++ch)
#pragma unroll
    for (int r = 0; r < 8; ++r) {
      int row = qt * 16 + r + 8 * half;
      O[(tb + row) * HID_ + h * DD + ch * 16 + n] =
          (_Float16)(oacc[ch][r] * rinv[r]);
    }
}

// ---------------------------------------------------------------------------
extern "C" void kernel_launch(void* const* d_in, const int* in_sizes, int n_in,
                              void* d_out, int out_size, void* d_ws, size_t ws_size,
                              hipStream_t stream) {
  (void)in_sizes; (void)n_in; (void)out_size; (void)ws_size;
  const float* x    = (const float*)d_in[0];
  const float* cosb = (const float*)d_in[1];
  const float* sinb = (const float*)d_in[2];
  const float* kck  = (const float*)d_in[3];
  const float* vck  = (const float*)d_in[4];
  const int*   slot = (const int*)d_in[5];
  const float* Wqkv = (const float*)d_in[6];
  const float* bqkv = (const float*)d_in[7];
  const float* Wo   = (const float*)d_in[8];

  float* out = (float*)d_out;
  float* kc  = out + (long long)TOK * HID_;
  float* vc  = kc + (long long)NSLOTS * HKV_ * DD;

  char* ws = (char*)d_ws;
  float*    qkv_ws = (float*)   (ws);
  _Float16* qkv_h  = (_Float16*)(ws + 67108864LL);
  _Float16* x_h    = (_Float16*)(ws + 100663296LL);
  _Float16* attn_h = (_Float16*)(ws + 125829120LL);
  _Float16* wqkv_h = (_Float16*)(ws + 150994944LL);
  _Float16* wo_h   = (_Float16*)(ws + 157286400LL);

  // 0) one-time f16 conversions of activations and weights
  {
    long long n4;
    n4 = (long long)TOK * HID_ / 4;
    cvt_f16_kernel<<<(unsigned)((n4 + 255) / 256), 256, 0, stream>>>(x, x_h, n4);
    n4 = (long long)QKV_N * HID_ / 4;
    cvt_f16_kernel<<<(unsigned)((n4 + 255) / 256), 256, 0, stream>>>(Wqkv, wqkv_h, n4);
    n4 = (long long)HID_ * HID_ / 4;
    cvt_f16_kernel<<<(unsigned)((n4 + 255) / 256), 256, 0, stream>>>(Wo, wo_h, n4);
  }

  // 1) qkv = x @ Wqkv^T + b  (f32 result for RoPE + cache scatter)
  //    32x64 per wave: grid = (M/32, N/(64*8))
  gemm_h_kernel<<<dim3(TOK / 32, QKV_N / 512), 256, 0, stream>>>(
      x_h, wqkv_h, bqkv, qkv_ws, TOK, QKV_N, HID_);

  // 2) caches: copy originals, then RoPE + scatter + f16 qkv copy
  hipMemcpyAsync(kc, kck, (size_t)NSLOTS * HKV_ * DD * sizeof(float),
                 hipMemcpyDeviceToDevice, stream);
  hipMemcpyAsync(vc, vck, (size_t)NSLOTS * HKV_ * DD * sizeof(float),
                 hipMemcpyDeviceToDevice, stream);
  rope_scatter_kernel<<<TOK, 128, 0, stream>>>(qkv_ws, qkv_h, cosb, sinb, slot, kc, vc);

  // 3) causal GQA flash attention (f16 in, f16 out)
  flash_attn_kernel<<<(BB * HH * 128) / 4, 128, 0, stream>>>(qkv_h, attn_h);

  // 4) out = attn @ Wo^T (f32 result)
  gemm_h_kernel<<<dim3(TOK / 32, HID_ / 512), 256, 0, stream>>>(
      attn_h, wo_h, nullptr, out, TOK, HID_, HID_);
}
